// GTSMambaBottleneck_84129819394232
// MI455X (gfx1250) — matrix operand order repaired and checked by reading, not compiled, not run on hardware
//
#include <hip/hip_runtime.h>

typedef __bf16 bf16;
typedef __attribute__((ext_vector_type(16))) __bf16 v16bf;
typedef __attribute__((ext_vector_type(8)))  float  v8f;
typedef __attribute__((ext_vector_type(4)))  float  v4f;
typedef __attribute__((ext_vector_type(2)))  float  v2f;

// ---------------------------------------------------------------------------
// WMMA helpers (CDNA5 wave32, v_wmma_f32_16x16x32_bf16)
// ---------------------------------------------------------------------------
static __device__ __forceinline__ v8f wmma_bf16(v16bf a, v16bf b, v8f c) {
  return __builtin_amdgcn_wmma_f32_16x16x32_bf16(false, a, false, b, (short)0, c,
                                                 false, false);
}

// Operands staged in LDS pre-swizzled in fragment order: 32B/lane contiguous.
static __device__ __forceinline__ v16bf load_frag(const bf16* sw) {
  return *(const v16bf*)(sw + ((threadIdx.x & 31) << 4));
}

// A-matrix element (row m, col k) of a 16x32 tile -> fragment order
static __device__ __forceinline__ void storeAelem(bf16* frag, int m, int k, float v) {
  const int ln = m + (((k >> 3) & 1) << 4);
  const int el = (k & 7) | ((k >> 4) << 3);
  frag[ln * 16 + el] = (bf16)v;
}

// B-matrix element (row k, col n) of a 32x16 tile -> fragment order
static __device__ __forceinline__ void storeBelem(bf16* frag, int k, int n, float v) {
  frag[(n + (((k >> 3) & 1) << 4)) * 16 + ((k & 7) | ((k >> 4) << 3))] = (bf16)v;
}

// C/D fragment: VGPR r holds row (r + 8*(lane>=16)), col = lane&15
static __device__ __forceinline__ void store_C16x16(float* sm, int ldr, v8f c) {
  const int lane = threadIdx.x & 31;
  const int n    = lane & 15;
  const int mb   = (lane < 16) ? 0 : 8;
#pragma unroll
  for (int r = 0; r < 8; ++r) sm[(mb + r) * ldr + n] = c[r];
}

static __device__ __forceinline__ float fast_rcp(float x) {
  return __builtin_amdgcn_rcpf(x);
}
static __device__ __forceinline__ float sigmoid_f(float v) {
  return fast_rcp(1.f + __expf(-v));
}
static __device__ __forceinline__ float silu_f(float v) {
  return v * sigmoid_f(v);
}

// (seq n, token l, absolute channel c) -> flat index into (B=2,C=96,D=16,H=32,W=32)
static __device__ __forceinline__ size_t xidx(int g, int n, int l, int c) {
  int b, d_, h_, w_;
  if (g == 0)      { b = n >> 4;  d_ = n & 15;        h_ = l >> 5;        w_ = l & 31; }
  else if (g == 1) { b = n >> 10; h_ = (n >> 5) & 31; w_ = n & 31;        d_ = l;      }
  else             { b = n;       d_ = l >> 10;       h_ = (l >> 5) & 31; w_ = l & 31; }
  return ((size_t)(b * 96 + c) * 16 + (size_t)d_) * 1024 + (size_t)(h_ * 32 + w_);
}

// LayerNorm of one token held lane-per-channel across a wave
static __device__ __forceinline__ float ln_lane(float v, const float* g_, const float* b_) {
  const int lane = threadIdx.x & 31;
  float s = v;
#pragma unroll
  for (int off = 16; off; off >>= 1) s += __shfl_xor(s, off, 32);
  const float mu = s * (1.f / 32.f);
  const float dv = v - mu;
  float q = dv * dv;
#pragma unroll
  for (int off = 16; off; off >>= 1) q += __shfl_xor(q, off, 32);
  const float rs = rsqrtf(q * (1.f / 32.f) + 1e-5f);
  return dv * rs * g_[lane] + b_[lane];
}

// ---------------------------------------------------------------------------
// Mamba chunked pipeline: one workgroup per 64-token superchunk (4x16 inner).
// phase 0 (SUM):   compute P = prod(dA), h_loc over superchunk (from h=0)
// phase 1 (APPLY): full pipeline seeded with exact h_start; write out_pre
// ---------------------------------------------------------------------------
struct MambaSmem {
  __attribute__((aligned(32))) bf16 w_inB[8][32][16];     // in-proj B tiles
  __attribute__((aligned(32))) bf16 w_xpB[2][3][32][16];  // x-proj B tiles (cols: B|C|dt)
  __attribute__((aligned(32))) bf16 w_outB[2][2][32][16]; // out-proj B tiles
  __attribute__((aligned(32))) bf16 hA[32][16];           // LN'd tokens (A frag)
  __attribute__((aligned(32))) bf16 xsA[2][32][16];       // silu(conv) (A frags)
  __attribute__((aligned(32))) bf16 yA[2][32][16];        // gated y (A frags)
  __attribute__((aligned(16))) float xz[16][128];         // x [0..63], z [64..127]
  __attribute__((aligned(16))) float dbc[16][48];         // B[0..15] C[16..31] dt[32..33]
  __attribute__((aligned(8)))  float dtxs[16][64][2];     // {softplus dt, silu xs}
  bf16  w_in_x[64][32];   // raw x-half of w_in (for conv-history matvec)
  bf16  hHist[3][32];     // LN'd history tokens
  float xhist[3][64];
  float w_dt[64][2];
  float b_dt[64];
  float Aneg[64][16];
  float Dp[64];
  float conv_w[64][4];
  float conv_b[64];
  float ln_g[32], ln_b[32];
};

__global__ __launch_bounds__(256) void gts_mamba_chunk_kernel(
    const float* __restrict__ x,
    const float* __restrict__ ln_g, const float* __restrict__ ln_b,
    const float* __restrict__ w_in, const float* __restrict__ conv_w,
    const float* __restrict__ conv_b, const float* __restrict__ w_xp,
    const float* __restrict__ w_dt, const float* __restrict__ b_dt,
    const float* __restrict__ A_log, const float* __restrict__ Dp,
    const float* __restrict__ w_out,
    float* __restrict__ out_pre, int g, int L, int phase,
    const float* __restrict__ hstart,
    float* __restrict__ Psum, float* __restrict__ Hsum) {
  __shared__ MambaSmem sm;
  const int tid  = threadIdx.x;
  const int wave = tid >> 5;
  const int lane = tid & 31;
  const int n    = blockIdx.y;             // sequence
  const int sc   = blockIdx.x;             // superchunk (64 tokens)
  const bool is_sum = (phase == 0);
  int nsub = (L >> 4) - (sc << 2);         // 16-token chunks in this superchunk
  if (nsub > 4) nsub = 4;

  // ---- stage weights (fragment-ordered bf16) ----
  for (int i = tid; i < 128 * 32; i += 256) {
    const int o = i >> 5, c = i & 31;
    const float v = w_in[g * 4096 + i];
    storeBelem(&sm.w_inB[o >> 4][0][0], c, o & 15, v);
    if (o < 64) sm.w_in_x[o][c] = (bf16)v;
  }
  for (int i = tid; i < 2 * 3 * 512; i += 256) (&sm.w_xpB[0][0][0][0])[i] = (bf16)0.f;
  __syncthreads();
  for (int i = tid; i < 34 * 64; i += 256) {
    const int j = i >> 6, d = i & 63;
    const int jn = (j < 2) ? (32 + j) : (j - 2);   // permute: B->0..15, C->16..31, dt->32..33
    storeBelem(&sm.w_xpB[d >> 5][jn >> 4][0][0], d & 31, jn & 15,
               w_xp[g * 34 * 64 + j * 64 + d]);
  }
  for (int i = tid; i < 2048; i += 256) {
    const int o = i >> 6, d = i & 63;
    storeBelem(&sm.w_outB[d >> 5][o >> 4][0][0], d & 31, o & 15, w_out[g * 2048 + i]);
  }
  for (int i = tid; i < 64 * 16; i += 256)
    sm.Aneg[i >> 4][i & 15] = -__expf(A_log[g * 1024 + i]);
  if (tid < 128) sm.w_dt[tid >> 1][tid & 1] = w_dt[g * 128 + tid];
  if (tid < 64) {
    sm.b_dt[tid]   = b_dt[g * 64 + tid];
    sm.Dp[tid]     = Dp[g * 64 + tid];
    sm.conv_b[tid] = conv_b[g * 64 + tid];
  }
  for (int i = tid; i < 64 * 4; i += 256) sm.conv_w[i >> 2][i & 3] = conv_w[g * 256 + i];
  if (tid < 32) { sm.ln_g[tid] = ln_g[g * 32 + tid]; sm.ln_b[tid] = ln_b[g * 32 + tid]; }
  __syncthreads();

  // ---- conv history for superchunk start (recomputed locally, exact) ----
  const int lh0 = sc << 6;
  if (lh0 == 0) {
    if (tid < 192) sm.xhist[tid / 64][tid % 64] = 0.f;
  } else {
    if (wave < 3) {   // LN of tokens lh0-3..lh0-1 (lane = channel)
      const float v = x[xidx(g, n, lh0 - 3 + wave, g * 32 + lane)];
      sm.hHist[wave][lane] = (bf16)ln_lane(v, sm.ln_g, sm.ln_b);
    }
    __syncthreads();
    if (tid < 192) {  // history xm = hHist @ w_in_x^T
      const int th = tid / 64, d = tid % 64;
      float acc = 0.f;
#pragma unroll 8
      for (int c = 0; c < 32; ++c)
        acc += (float)sm.hHist[th][c] * (float)sm.w_in_x[d][c];
      sm.xhist[th][d] = acc;
    }
  }

  // ---- scan state: thread owns d_own = tid>>2, states (tid&3)*4 .. +3 ----
  const int d_own  = tid >> 2;
  const int s_base = (tid & 3) * 4;
  v4f hstate = {0.f, 0.f, 0.f, 0.f};
  if (!is_sum && hstart)
    hstate = *(const v4f*)&hstart[((size_t)n * gridDim.x + sc) * 1024 + tid * 4];
  v4f Pp = {1.f, 1.f, 1.f, 1.f};
  v4f Ad;
#pragma unroll
  for (int j = 0; j < 4; ++j) Ad[j] = sm.Aneg[d_own][s_base + j];

  for (int ck = 0; ck < nsub; ++ck) {
    const int l0 = lh0 + (ck << 4);
    __syncthreads();

    // ---- load 16 tokens + LayerNorm (lane = channel, 2 tokens per wave) ----
#pragma unroll
    for (int rep = 0; rep < 2; ++rep) {
      const int t = wave + rep * 8;
      const int l = l0 + t;
      const float v = x[xidx(g, n, l, g * 32 + lane)];
      if (ck + 1 < nsub) __builtin_prefetch(&x[xidx(g, n, l + 16, g * 32 + lane)], 0, 1);
      storeAelem(&sm.hA[0][0], t, lane, ln_lane(v, sm.ln_g, sm.ln_b));
    }
    __syncthreads();

    // ---- in-projection: (16x32) @ (32x128); SUM phase skips z tiles ----
    if (!is_sum || wave < 4) {
      v16bf a  = load_frag(&sm.hA[0][0]);
      v16bf bb = load_frag(&sm.w_inB[wave][0][0]);
      v8f c = {};
      c = wmma_bf16(a, bb, c);
      store_C16x16(&sm.xz[0][wave * 16], 128, c);
    }
    __syncthreads();

    // ---- causal depthwise conv K=4 + SiLU ----
#pragma unroll
    for (int i = 0; i < 4; ++i) {
      const int p = tid * 4 + i;
      const int t = p >> 6, d = p & 63;
      float acc = sm.conv_b[d];
#pragma unroll
      for (int k = 0; k < 4; ++k) {
        const int tt = t - 3 + k;
        const float xv = (tt >= 0) ? sm.xz[tt][d] : sm.xhist[3 + tt][d];
        acc += sm.conv_w[d][k] * xv;
      }
      const float xsv = silu_f(acc);
      sm.dtxs[t][d][1] = xsv;
      storeAelem(&sm.xsA[d >> 5][0][0], t, d & 31, xsv);
    }
    __syncthreads();
    if (tid < 192) sm.xhist[tid / 64][tid % 64] = sm.xz[13 + tid / 64][tid % 64];

    // ---- x-projection: (16x64) @ (64x48); SUM skips the C tile (wave 1) ----
    if (wave < 3 && !(is_sum && wave == 1)) {
      v16bf a0 = load_frag(&sm.xsA[0][0][0]);
      v16bf a1 = load_frag(&sm.xsA[1][0][0]);
      v16bf b0 = load_frag(&sm.w_xpB[0][wave][0][0]);
      v16bf b1 = load_frag(&sm.w_xpB[1][wave][0][0]);
      v8f c = {};
      c = wmma_bf16(a0, b0, c);
      c = wmma_bf16(a1, b1, c);
      store_C16x16(&sm.dbc[0][wave * 16], 48, c);
    }
    __syncthreads();

    // ---- dt = softplus(dt_raw @ w_dt^T + b_dt) ----
#pragma unroll
    for (int i = 0; i < 4; ++i) {
      const int p = tid * 4 + i;
      const int t = p >> 6, d = p & 63;
      const float v = sm.dbc[t][32] * sm.w_dt[d][0] + sm.dbc[t][33] * sm.w_dt[d][1] + sm.b_dt[d];
      sm.dtxs[t][d][0] = (v > 20.f) ? v : __logf(1.f + __expf(v));
    }
    __syncthreads();

    // ---- sequential SSM scan over 16 tokens ----
    for (int t = 0; t < 16; ++t) {
      const v2f dxv = *(const v2f*)&sm.dtxs[t][d_own][0];  // {dt, xs}
      const float dtv = dxv[0], xsv = dxv[1];
      const float dx  = dtv * xsv;
      const v4f Bv = *(const v4f*)&sm.dbc[t][s_base];
      float acc = 0.f;
      if (is_sum) {
#pragma unroll
        for (int j = 0; j < 4; ++j) {
          const float dA = __expf(dtv * Ad[j]);
          Pp[j] *= dA;
          hstate[j] = dA * hstate[j] + dx * Bv[j];
        }
      } else {
        const v4f Cv = *(const v4f*)&sm.dbc[t][16 + s_base];
#pragma unroll
        for (int j = 0; j < 4; ++j) {
          const float dA = __expf(dtv * Ad[j]);
          hstate[j] = dA * hstate[j] + dx * Bv[j];
          acc += hstate[j] * Cv[j];
        }
        acc += __shfl_xor(acc, 1, 32);
        acc += __shfl_xor(acc, 2, 32);
        if ((tid & 3) == 0) {
          float yv = acc + sm.Dp[d_own] * xsv;
          yv *= silu_f(sm.xz[t][64 + d_own]);
          storeAelem(&sm.yA[d_own >> 5][0][0], t, d_own & 31, yv);
        }
      }
    }

    // ---- out-projection: (16x64) @ (64x32); 2 N-tiles, waves 0..1 ----
    if (!is_sum) {
      __syncthreads();
      if (wave < 2) {
        v16bf a0 = load_frag(&sm.yA[0][0][0]);
        v16bf a1 = load_frag(&sm.yA[1][0][0]);
        v16bf b0 = load_frag(&sm.w_outB[0][wave][0][0]);
        v16bf b1 = load_frag(&sm.w_outB[1][wave][0][0]);
        v8f c = {};
        c = wmma_bf16(a0, b0, c);
        c = wmma_bf16(a1, b1, c);
        const int oc = wave * 16 + (lane & 15);
        const int mb = (lane < 16) ? 0 : 8;
#pragma unroll
        for (int r = 0; r < 8; ++r)
          out_pre[xidx(g, n, l0 + mb + r, g * 32 + oc)] = c[r];
      }
    }
  }

  if (is_sum) {
    const size_t off = ((size_t)n * gridDim.x + sc) * 1024 + tid * 4;
    *(v4f*)&Psum[off] = Pp;
    *(v4f*)&Hsum[off] = hstate;
  }
}

// Sequential combine of superchunk summaries -> exact h_start per superchunk.
__global__ __launch_bounds__(256) void gts_combine_kernel(
    const float* __restrict__ Psum, const float* __restrict__ Hsum,
    float* __restrict__ hstart, int nsuper) {
  const int n   = blockIdx.x;
  const int idx = threadIdx.x * 4;
  v4f h = {0.f, 0.f, 0.f, 0.f};
  for (int sc = 0; sc < nsuper; ++sc) {
    const size_t off = ((size_t)n * nsuper + sc) * 1024 + idx;
    *(v4f*)&hstart[off] = h;
    const v4f P  = *(const v4f*)&Psum[off];
    const v4f Hl = *(const v4f*)&Hsum[off];
    h = P * h + Hl;
  }
}

// ---------------------------------------------------------------------------
// per-(b,c) mean over 16384 voxels (for ECA)
// ---------------------------------------------------------------------------
__global__ __launch_bounds__(256) void gts_mean_kernel(const float* __restrict__ out_pre,
                                                       float* __restrict__ smean) {
  const int bc = blockIdx.x;
  const float* p = out_pre + (size_t)bc * 16384;
  float s = 0.f;
  for (int i = threadIdx.x; i < 16384; i += 256) s += p[i];
  __shared__ float red[8];
#pragma unroll
  for (int off = 16; off; off >>= 1) s += __shfl_xor(s, off, 32);
  if ((threadIdx.x & 31) == 0) red[threadIdx.x >> 5] = s;
  __syncthreads();
  if (threadIdx.x < 8) {
    float v = red[threadIdx.x];
#pragma unroll
    for (int off = 4; off; off >>= 1) v += __shfl_xor(v, off, 32);
    if (threadIdx.x == 0) smean[bc] = v * (1.f / 16384.f);
  }
}

__global__ void gts_eca_kernel(const float* __restrict__ s,
                               const float* __restrict__ eca_w,
                               float* __restrict__ a) {
  const int i = blockIdx.x * blockDim.x + threadIdx.x;
  if (i >= 192) return;
  const int c = i % 96;
  const float left  = (c > 0)  ? s[i - 1] : 0.f;
  const float right = (c < 95) ? s[i + 1] : 0.f;
  a[i] = sigmoid_f(eca_w[0] * left + eca_w[1] * s[i] + eca_w[2] * right);
}

// ---------------------------------------------------------------------------
// fuse 96x96 GEMM + ECA scale + instance-norm statistics
// ---------------------------------------------------------------------------
struct FuseSmem {
  __attribute__((aligned(32))) bf16 A[4][3][32][16];
  __attribute__((aligned(32))) bf16 W[3][6][32][16];
  float aScale[96];
};

__global__ __launch_bounds__(256) void gts_fuse_kernel(
    const float* __restrict__ out_pre, const float* __restrict__ a,
    const float* __restrict__ fuse_w, float* __restrict__ f,
    float* __restrict__ insum, float* __restrict__ insum2) {
  __shared__ FuseSmem sm;
  const int tid  = threadIdx.x;
  const int wave = tid >> 5;
  const int lane = tid & 31;
  const int v0   = blockIdx.x * 64;
  const int b    = v0 >> 14;
  const int vb   = v0 & 16383;

  for (int i = tid; i < 96 * 96; i += 256) {
    const int o = i / 96, c = i % 96;
    storeBelem(&sm.W[c >> 5][o >> 4][0][0], c & 31, o & 15, fuse_w[i]);
  }
  if (tid < 96) sm.aScale[tid] = a[b * 96 + tid];
  __syncthreads();
  for (int i = tid; i < 64 * 96; i += 256) {
    const int r = i / 96, c = i % 96;
    const float v = out_pre[(size_t)(b * 96 + c) * 16384 + vb + r] * sm.aScale[c];
    storeAelem(&sm.A[r >> 4][c >> 5][0][0], r & 15, c & 31, v);
  }
  __syncthreads();

  for (int j = wave; j < 24; j += 8) {
    const int mt = j / 6, nt = j % 6;
    v8f c = {};
#pragma unroll
    for (int kk = 0; kk < 3; ++kk) {
      v16bf af = load_frag(&sm.A[mt][kk][0][0]);
      v16bf bf = load_frag(&sm.W[kk][nt][0][0]);
      c = wmma_bf16(af, bf, c);
    }
    const int oc = nt * 16 + (lane & 15);
    const int mb = (lane < 16) ? 0 : 8;
    float lsum = 0.f, lsq = 0.f;
#pragma unroll
    for (int r = 0; r < 8; ++r) {
      const float v = c[r];
      f[(size_t)(b * 96 + oc) * 16384 + vb + mt * 16 + mb + r] = v;
      lsum += v;
      lsq  += v * v;
    }
    lsum += __shfl_xor(lsum, 16, 32);
    lsq  += __shfl_xor(lsq, 16, 32);
    if (lane < 16) {
      atomicAdd(&insum[b * 96 + oc], lsum);
      atomicAdd(&insum2[b * 96 + oc], lsq);
    }
  }
}

__global__ __launch_bounds__(256) void gts_final_kernel(
    const float* __restrict__ f, const float* __restrict__ insum,
    const float* __restrict__ insum2, const float* __restrict__ x,
    float* __restrict__ out) {
  const int i = blockIdx.x * 256 + threadIdx.x;
  if (i >= 3145728) return;
  const int bo = i >> 14;
  const float mu  = insum[bo] * (1.f / 16384.f);
  const float var = insum2[bo] * (1.f / 16384.f) - mu * mu;
  const float v = (f[i] - mu) * rsqrtf(var + 1e-5f);
  out[i] = silu_f(v) + x[i];
}

__global__ void gts_zero_kernel(float* __restrict__ p, int n) {
  const int i = blockIdx.x * blockDim.x + threadIdx.x;
  if (i < n) p[i] = 0.f;
}

// ---------------------------------------------------------------------------
extern "C" void kernel_launch(void* const* d_in, const int* in_sizes, int n_in,
                              void* d_out, int out_size, void* d_ws, size_t ws_size,
                              hipStream_t stream) {
  const float* x      = (const float*)d_in[0];
  const float* ln_g   = (const float*)d_in[1];
  const float* ln_b   = (const float*)d_in[2];
  const float* w_in   = (const float*)d_in[3];
  const float* conv_w = (const float*)d_in[4];
  const float* conv_b = (const float*)d_in[5];
  const float* w_xp   = (const float*)d_in[6];
  const float* w_dt   = (const float*)d_in[7];
  const float* b_dt   = (const float*)d_in[8];
  const float* A_log  = (const float*)d_in[9];
  const float* Dp     = (const float*)d_in[10];
  const float* w_out  = (const float*)d_in[11];
  const float* eca_w  = (const float*)d_in[12];
  const float* fuse_w = (const float*)d_in[13];

  float* ws      = (float*)d_ws;
  float* out_pre = ws;                 // 3145728 floats
  float* fbuf    = ws + 3145728;       // 3145728 floats
  float* smean   = ws + 6291456;       // 192
  float* aatt    = ws + 6291648;       // 192
  float* insum   = ws + 6291840;       // 192
  float* insum2  = ws + 6292032;       // 192
  float* Psum    = ws + 6292224;       // 524288 (max nseq*nsuper*1024)
  float* Hsum    = ws + 6816512;       // 524288
  float* hstart  = ws + 7340800;       // 524288

  gts_zero_kernel<<<2, 256, 0, stream>>>(insum, 384);

#define MAMBA_ARGS x, ln_g, ln_b, w_in, conv_w, conv_b, w_xp, w_dt, b_dt, A_log, Dp, w_out, out_pre

  // group 0: 32 seqs, L=1024 -> 16 superchunks
  gts_mamba_chunk_kernel<<<dim3(16, 32), 256, 0, stream>>>(MAMBA_ARGS, 0, 1024, 0,
                                                           nullptr, Psum, Hsum);
  gts_combine_kernel<<<32, 256, 0, stream>>>(Psum, Hsum, hstart, 16);
  gts_mamba_chunk_kernel<<<dim3(16, 32), 256, 0, stream>>>(MAMBA_ARGS, 0, 1024, 1,
                                                           hstart, Psum, Hsum);

  // group 1: 2048 seqs, L=16 -> single chunk, h_start = 0 (apply only)
  gts_mamba_chunk_kernel<<<dim3(1, 2048), 256, 0, stream>>>(MAMBA_ARGS, 1, 16, 1,
                                                            nullptr, Psum, Hsum);

  // group 2: 2 seqs, L=16384 -> 256 superchunks
  gts_mamba_chunk_kernel<<<dim3(256, 2), 256, 0, stream>>>(MAMBA_ARGS, 2, 16384, 0,
                                                           nullptr, Psum, Hsum);
  gts_combine_kernel<<<2, 256, 0, stream>>>(Psum, Hsum, hstart, 256);
  gts_mamba_chunk_kernel<<<dim3(256, 2), 256, 0, stream>>>(MAMBA_ARGS, 2, 16384, 1,
                                                           hstart, Psum, Hsum);

  gts_mean_kernel<<<192, 256, 0, stream>>>(out_pre, smean);
  gts_eca_kernel<<<1, 256, 0, stream>>>(smean, eca_w, aatt);
  gts_fuse_kernel<<<512, 256, 0, stream>>>(out_pre, aatt, fuse_w, fbuf, insum, insum2);
  gts_final_kernel<<<12288, 256, 0, stream>>>(fbuf, insum, insum2, x, (float*)d_out);
#undef MAMBA_ARGS
}
